// CausalSelfAttention_19516331393504
// MI455X (gfx1250) — matrix operand order repaired and checked by reading, not compiled, not run on hardware
//
#include <hip/hip_runtime.h>
#include <hip/hip_bf16.h>

typedef __attribute__((ext_vector_type(16))) _Float16 v16h;
typedef __attribute__((ext_vector_type(8)))  _Float16 v8h;
typedef __attribute__((ext_vector_type(8)))  float    v8f;

union V16U { v16h v; v8h h[2]; };

// ---- WMMA helpers -----------------------------------------------------------

__device__ __forceinline__ v8f wmma16(v16h a, v16h b, v8f c) {
  return __builtin_amdgcn_wmma_f32_16x16x32_f16(
      /*neg_a=*/false, a, /*neg_b=*/false, b,
      /*c_mod=*/(short)0, c, /*reuse_a=*/false, /*reuse_b=*/false);
}

// A-fragment of a 16x32 f16 tile stored row-major with row stride `stride`
// (halves). Lanes 0-15: row m holds K 0..15;
// lanes 16-31: same row m, K 16..31.
__device__ __forceinline__ v16h load_a16(const _Float16* base, int stride, int lane) {
  const int m    = lane & 15;
  const int koff = (lane & 16) ? 16 : 0;
  const _Float16* p = base + (size_t)m * stride + koff;
  V16U u;
  u.h[0] = *(const v8h*)(p);
  u.h[1] = *(const v8h*)(p + 8);
  return u.v;
}

// B-fragment of a 32x16 f16 tile stored COLUMN-contiguous: column n is a
// contiguous run of K halves at base + n*stride. Lanes 0-15 take K 0..15,
// lanes 16-31 take K 16..31.
__device__ __forceinline__ v16h load_bt16(const _Float16* base, int stride, int lane) {
  const int n    = lane & 15;
  const int koff = (lane & 16) ? 16 : 0;
  return *(const v16h*)(base + (size_t)n * stride + koff);
}

// ---- Conversion kernels -----------------------------------------------------

__global__ void cvt_f16(const float* __restrict__ in, _Float16* __restrict__ out, int n) {
  int i = blockIdx.x * 256 + threadIdx.x;
  if (i < n) out[i] = (_Float16)in[i];
}

// in: [K][N] row-major f32 ; out: [N][K] row-major f16 (i.e. transposed)
__global__ void cvt_t_f16(const float* __restrict__ in, _Float16* __restrict__ out,
                          int K, int N) {
  int i = blockIdx.x * 256 + threadIdx.x;
  if (i < K * N) {
    int k = i / N, n = i % N;
    out[(size_t)n * K + k] = (_Float16)in[i];
  }
}

// ---- QKV projection GEMM: [8192,1024] x [1024,3072] + b ---------------------
// Block = 128 threads (4 waves). Wave w: rows m0 = bx*128 + w*32 (32-row tile,
// two 16-row WMMA subtiles sharing B fragments), cols n0 = by*64.
// Double-buffered K loop: fragments for k+32 are loaded before the WMMAs for k.
// Epilogue scatters into q[bh][t][d], k[bh][t][d], vT[bh][d][t] as f16.

__global__ __launch_bounds__(128) void gemm_qkv(
    const _Float16* __restrict__ xh,   // [8192][1024]
    const _Float16* __restrict__ wqT,  // [3072][1024]  (w_qkv transposed)
    const float*    __restrict__ bq,   // [3072]
    _Float16* __restrict__ qh, _Float16* __restrict__ kh, _Float16* __restrict__ vt) {
  const int lane = threadIdx.x & 31;
  const int wave = threadIdx.x >> 5;
  const int m0 = blockIdx.x * 128 + wave * 32;
  const int n0 = blockIdx.y * 64;

  v8f acc[2][4] = {};
  const _Float16* arow0 = xh + (size_t)m0 * 1024;
  const _Float16* arow1 = arow0 + 16 * 1024;
  const _Float16* bbase = wqT + (size_t)n0 * 1024;

  // prologue: fragments for k = 0
  v16h a0 = load_a16(arow0, 1024, lane);
  v16h a1 = load_a16(arow1, 1024, lane);
  v16h b0 = load_bt16(bbase + 0 * 16384, 1024, lane);
  v16h b1 = load_bt16(bbase + 1 * 16384, 1024, lane);
  v16h b2 = load_bt16(bbase + 2 * 16384, 1024, lane);
  v16h b3 = load_bt16(bbase + 3 * 16384, 1024, lane);

  for (int k0 = 0; k0 < 1024; k0 += 64) {
    // issue loads for k0+32 before consuming k0 fragments
    const int k1 = k0 + 32;
    v16h a0n = load_a16(arow0 + k1, 1024, lane);
    v16h a1n = load_a16(arow1 + k1, 1024, lane);
    v16h b0n = load_bt16(bbase + 0 * 16384 + k1, 1024, lane);
    v16h b1n = load_bt16(bbase + 1 * 16384 + k1, 1024, lane);
    v16h b2n = load_bt16(bbase + 2 * 16384 + k1, 1024, lane);
    v16h b3n = load_bt16(bbase + 3 * 16384 + k1, 1024, lane);
    __builtin_prefetch(arow0 + k0 + 256, 0, 1);

    acc[0][0] = wmma16(a0, b0, acc[0][0]);
    acc[1][0] = wmma16(a1, b0, acc[1][0]);
    acc[0][1] = wmma16(a0, b1, acc[0][1]);
    acc[1][1] = wmma16(a1, b1, acc[1][1]);
    acc[0][2] = wmma16(a0, b2, acc[0][2]);
    acc[1][2] = wmma16(a1, b2, acc[1][2]);
    acc[0][3] = wmma16(a0, b3, acc[0][3]);
    acc[1][3] = wmma16(a1, b3, acc[1][3]);

    // issue loads for k0+64 (clamped in-bounds dummy on the final pass)
    const int k2 = (k0 + 64 < 1024) ? k0 + 64 : 0;
    a0 = load_a16(arow0 + k2, 1024, lane);
    a1 = load_a16(arow1 + k2, 1024, lane);
    b0 = load_bt16(bbase + 0 * 16384 + k2, 1024, lane);
    b1 = load_bt16(bbase + 1 * 16384 + k2, 1024, lane);
    b2 = load_bt16(bbase + 2 * 16384 + k2, 1024, lane);
    b3 = load_bt16(bbase + 3 * 16384 + k2, 1024, lane);

    acc[0][0] = wmma16(a0n, b0n, acc[0][0]);
    acc[1][0] = wmma16(a1n, b0n, acc[1][0]);
    acc[0][1] = wmma16(a0n, b1n, acc[0][1]);
    acc[1][1] = wmma16(a1n, b1n, acc[1][1]);
    acc[0][2] = wmma16(a0n, b2n, acc[0][2]);
    acc[1][2] = wmma16(a1n, b2n, acc[1][2]);
    acc[0][3] = wmma16(a0n, b3n, acc[0][3]);
    acc[1][3] = wmma16(a1n, b3n, acc[1][3]);
  }

  const int rshift = (lane & 16) ? 8 : 0;
  const int nl = lane & 15;
#pragma unroll
  for (int mi = 0; mi < 2; ++mi) {
#pragma unroll
    for (int j = 0; j < 4; ++j) {
      const int col = n0 + 16 * j + nl;
      const float bias = bq[col];
#pragma unroll
      for (int r = 0; r < 8; ++r) {
        const int row = m0 + 16 * mi + r + rshift;
        const _Float16 hv = (_Float16)(acc[mi][j][r] + bias);
        const int b = row >> 11, t = row & 2047;
        if (col < 1024) {
          const int h = col >> 6, d = col & 63;
          qh[((size_t)((b * 16 + h) * 2048 + t) << 6) + d] = hv;
        } else if (col < 2048) {
          const int f = col - 1024, h = f >> 6, d = f & 63;
          kh[((size_t)((b * 16 + h) * 2048 + t) << 6) + d] = hv;
        } else {
          const int f = col - 2048, h = f >> 6, d = f & 63;
          vt[((size_t)((b * 16 + h) * 64 + d) << 11) + t] = hv;   // transposed
        }
      }
    }
  }
}

// ---- Flash-attention kernel -------------------------------------------------
// One wave per (b,h, 16-query tile). Online softmax, 32 keys per step.
// V fragments are loaded before the softmax block so their latency hides
// under the exp/reduction VALU work; next key block is prefetched.

__global__ __launch_bounds__(128) void attn_kernel(
    const _Float16* __restrict__ qh,  // [64][2048][64]
    const _Float16* __restrict__ kh,  // [64][2048][64]
    const _Float16* __restrict__ vt,  // [64][64][2048]
    _Float16* __restrict__ atth) {    // [8192][1024]
  __shared__ __align__(64) _Float16 plds[4][16 * 32];

  const int lane = threadIdx.x & 31;
  const int wave = threadIdx.x >> 5;
  const int wid  = blockIdx.x * 4 + wave;
  const int qt = wid & 127;      // query tile in head
  const int bh = wid >> 7;       // 0..63 = b*16+h
  const int q0 = qt * 16;

  const _Float16* Q = qh + ((size_t)bh << 17);
  const _Float16* K = kh + ((size_t)bh << 17);
  const _Float16* V = vt + ((size_t)bh << 17);
  _Float16* myp = plds[wave];

  const v16h qa0 = load_a16(Q + (size_t)q0 * 64,      64, lane);
  const v16h qa1 = load_a16(Q + (size_t)q0 * 64 + 32, 64, lane);

  v8f o0 = {}, o1 = {}, o2 = {}, o3 = {};
  float mrun[8], lrun[8];
#pragma unroll
  for (int r = 0; r < 8; ++r) { mrun[r] = -3.0e38f; lrun[r] = 0.0f; }

  const int nl = lane & 15;
  const int rshift = (lane & 16) ? 8 : 0;

  for (int kb = 0; kb < q0 + 16; kb += 32) {
    // K fragments for this block
    v16h b00 = load_bt16(K + (size_t)kb * 64,             64, lane);
    v16h b01 = load_bt16(K + (size_t)kb * 64 + 32,        64, lane);
    v16h b10 = load_bt16(K + (size_t)(kb + 16) * 64,      64, lane);
    v16h b11 = load_bt16(K + (size_t)(kb + 16) * 64 + 32, 64, lane);
    // V fragments issued early: independent of softmax, consumed after it
    v16h vf0 = load_bt16(V + kb,             2048, lane);
    v16h vf1 = load_bt16(V + 16 * 2048 + kb, 2048, lane);
    v16h vf2 = load_bt16(V + 32 * 2048 + kb, 2048, lane);
    v16h vf3 = load_bt16(V + 48 * 2048 + kb, 2048, lane);
    __builtin_prefetch(K + (size_t)(kb + 32) * 64, 0, 1);

    // S tiles: 16 queries x 32 keys, K-dim = d = 64 (two WMMA each)
    v8f s0 = {}, s1 = {};
    s0 = wmma16(qa0, b00, s0);
    s0 = wmma16(qa1, b01, s0);
    s1 = wmma16(qa0, b10, s1);
    s1 = wmma16(qa1, b11, s1);

    // online softmax per row
#pragma unroll
    for (int r = 0; r < 8; ++r) {
      const int row = q0 + r + rshift;
      const int k0i = kb + nl;
      const int k1i = k0i + 16;
      float e0 = (k0i <= row) ? s0[r] * 0.125f : -1.0e30f;  // 1/sqrt(64)
      float e1 = (k1i <= row) ? s1[r] * 0.125f : -1.0e30f;
      float mx = fmaxf(e0, e1);
      mx = fmaxf(mx, __shfl_xor(mx, 1, 32));
      mx = fmaxf(mx, __shfl_xor(mx, 2, 32));
      mx = fmaxf(mx, __shfl_xor(mx, 4, 32));
      mx = fmaxf(mx, __shfl_xor(mx, 8, 32));
      const float mnew  = fmaxf(mrun[r], mx);
      const float alpha = __expf(mrun[r] - mnew);
      const float p0 = __expf(e0 - mnew);
      const float p1 = __expf(e1 - mnew);
      float sum = p0 + p1;
      sum += __shfl_xor(sum, 1, 32);
      sum += __shfl_xor(sum, 2, 32);
      sum += __shfl_xor(sum, 4, 32);
      sum += __shfl_xor(sum, 8, 32);
      lrun[r] = lrun[r] * alpha + sum;
      mrun[r] = mnew;
      o0[r] *= alpha; o1[r] *= alpha; o2[r] *= alpha; o3[r] *= alpha;
      const int rl = r + rshift;
      myp[rl * 32 + nl]      = (_Float16)p0;    // row-major 16x32 P tile
      myp[rl * 32 + 16 + nl] = (_Float16)p1;
    }
    // P (f16 A-fragment via LDS)  x  V (32 keys x 64 d), vT column-contiguous
    const v16h pa = load_a16(myp, 32, lane);
    o0 = wmma16(pa, vf0, o0);
    o1 = wmma16(pa, vf1, o1);
    o2 = wmma16(pa, vf2, o2);
    o3 = wmma16(pa, vf3, o3);
  }

  // normalize and store att output row-major [token][h*64+d] in f16
  const int b = bh >> 4, h = bh & 15;
#pragma unroll
  for (int r = 0; r < 8; ++r) {
    const float inv = 1.0f / lrun[r];
    const int t = q0 + r + rshift;
    const size_t rowbase = ((size_t)(b * 2048 + t) << 10) + (h << 6) + nl;
    atth[rowbase + 0]  = (_Float16)(o0[r] * inv);
    atth[rowbase + 16] = (_Float16)(o1[r] * inv);
    atth[rowbase + 32] = (_Float16)(o2[r] * inv);
    atth[rowbase + 48] = (_Float16)(o3[r] * inv);
  }
}

// ---- Output projection GEMM: [8192,1024] x [1024,1024] + b -> f32 -----------
// Same 32x64 wave tile + double-buffered K loop as gemm_qkv.

__global__ __launch_bounds__(128) void gemm_out(
    const _Float16* __restrict__ atth, // [8192][1024]
    const _Float16* __restrict__ woT,  // [1024][1024] (w_out transposed)
    const float*    __restrict__ bo,   // [1024]
    float* __restrict__ out) {         // [8192][1024]
  const int lane = threadIdx.x & 31;
  const int wave = threadIdx.x >> 5;
  const int m0 = blockIdx.x * 128 + wave * 32;
  const int n0 = blockIdx.y * 64;

  v8f acc[2][4] = {};
  const _Float16* arow0 = atth + (size_t)m0 * 1024;
  const _Float16* arow1 = arow0 + 16 * 1024;
  const _Float16* bbase = woT + (size_t)n0 * 1024;

  v16h a0 = load_a16(arow0, 1024, lane);
  v16h a1 = load_a16(arow1, 1024, lane);
  v16h b0 = load_bt16(bbase + 0 * 16384, 1024, lane);
  v16h b1 = load_bt16(bbase + 1 * 16384, 1024, lane);
  v16h b2 = load_bt16(bbase + 2 * 16384, 1024, lane);
  v16h b3 = load_bt16(bbase + 3 * 16384, 1024, lane);

  for (int k0 = 0; k0 < 1024; k0 += 64) {
    const int k1 = k0 + 32;
    v16h a0n = load_a16(arow0 + k1, 1024, lane);
    v16h a1n = load_a16(arow1 + k1, 1024, lane);
    v16h b0n = load_bt16(bbase + 0 * 16384 + k1, 1024, lane);
    v16h b1n = load_bt16(bbase + 1 * 16384 + k1, 1024, lane);
    v16h b2n = load_bt16(bbase + 2 * 16384 + k1, 1024, lane);
    v16h b3n = load_bt16(bbase + 3 * 16384 + k1, 1024, lane);
    __builtin_prefetch(arow0 + k0 + 256, 0, 1);

    acc[0][0] = wmma16(a0, b0, acc[0][0]);
    acc[1][0] = wmma16(a1, b0, acc[1][0]);
    acc[0][1] = wmma16(a0, b1, acc[0][1]);
    acc[1][1] = wmma16(a1, b1, acc[1][1]);
    acc[0][2] = wmma16(a0, b2, acc[0][2]);
    acc[1][2] = wmma16(a1, b2, acc[1][2]);
    acc[0][3] = wmma16(a0, b3, acc[0][3]);
    acc[1][3] = wmma16(a1, b3, acc[1][3]);

    const int k2 = (k0 + 64 < 1024) ? k0 + 64 : 0;
    a0 = load_a16(arow0 + k2, 1024, lane);
    a1 = load_a16(arow1 + k2, 1024, lane);
    b0 = load_bt16(bbase + 0 * 16384 + k2, 1024, lane);
    b1 = load_bt16(bbase + 1 * 16384 + k2, 1024, lane);
    b2 = load_bt16(bbase + 2 * 16384 + k2, 1024, lane);
    b3 = load_bt16(bbase + 3 * 16384 + k2, 1024, lane);

    acc[0][0] = wmma16(a0n, b0n, acc[0][0]);
    acc[1][0] = wmma16(a1n, b0n, acc[1][0]);
    acc[0][1] = wmma16(a0n, b1n, acc[0][1]);
    acc[1][1] = wmma16(a1n, b1n, acc[1][1]);
    acc[0][2] = wmma16(a0n, b2n, acc[0][2]);
    acc[1][2] = wmma16(a1n, b2n, acc[1][2]);
    acc[0][3] = wmma16(a0n, b3n, acc[0][3]);
    acc[1][3] = wmma16(a1n, b3n, acc[1][3]);
  }

  const int rshift = (lane & 16) ? 8 : 0;
  const int nl = lane & 15;
#pragma unroll
  for (int mi = 0; mi < 2; ++mi) {
#pragma unroll
    for (int j = 0; j < 4; ++j) {
      const int col = n0 + 16 * j + nl;
      const float bias = bo[col];
#pragma unroll
      for (int r = 0; r < 8; ++r) {
        const int row = m0 + 16 * mi + r + rshift;
        out[(size_t)row * 1024 + col] = acc[mi][j][r] + bias;
      }
    }
  }
}

// ---- Host launch ------------------------------------------------------------

extern "C" void kernel_launch(void* const* d_in, const int* in_sizes, int n_in,
                              void* d_out, int out_size, void* d_ws, size_t ws_size,
                              hipStream_t stream) {
  const float* x    = (const float*)d_in[0]; // [4,2048,1024]
  const float* wqkv = (const float*)d_in[1]; // [1024,3072]
  const float* bq   = (const float*)d_in[2]; // [3072]
  const float* wout = (const float*)d_in[3]; // [1024,1024]
  const float* bo   = (const float*)d_in[4]; // [1024]
  float* out = (float*)d_out;

  char* ws = (char*)d_ws;
  _Float16* xh  = (_Float16*)(ws + 0);          // 16 MB  x as f16
  _Float16* wqT = (_Float16*)(ws + 16777216);   //  6 MB  w_qkv^T f16
  _Float16* woT = (_Float16*)(ws + 23068672);   //  2 MB  w_out^T f16
  _Float16* qhp = (_Float16*)(ws + 25165824);   // 16 MB  q [bh][t][d]
  _Float16* khp = (_Float16*)(ws + 41943040);   // 16 MB  k [bh][t][d]
  _Float16* vtp = (_Float16*)(ws + 58720256);   // 16 MB  v^T [bh][d][t]
  _Float16* ath = (_Float16*)(ws + 75497472);   // 16 MB  att out f16

  cvt_f16 <<<(8388608 + 255) / 256, 256, 0, stream>>>(x, xh, 8388608);
  cvt_t_f16<<<(3145728 + 255) / 256, 256, 0, stream>>>(wqkv, wqT, 1024, 3072);
  cvt_t_f16<<<(1048576 + 255) / 256, 256, 0, stream>>>(wout, woT, 1024, 1024);

  gemm_qkv   <<<dim3(64, 48), 128, 0, stream>>>(xh, wqT, bq, qhp, khp, vtp);
  attn_kernel<<<2048,         128, 0, stream>>>(qhp, khp, vtp, ath);
  gemm_out   <<<dim3(64, 16), 128, 0, stream>>>(ath, woT, bo, out);
}
